// ReferenceAttention_54382875901987
// MI455X (gfx1250) — matrix operand order repaired and checked by reading, not compiled, not run on hardware
//
#include <hip/hip_runtime.h>
#include <hip/hip_bf16.h>
#include <cstdint>

// ---------------------------------------------------------------------------
// Qwen-style attention block for MI455X (gfx1250, wave32, WMMA).
//   1. convert hidden/W* to bf16 (workspace)
//   2. LDS-tiled WMMA GEMM (async global->LDS double buffering): Q/K/V proj
//   3. fused RMSNorm + RoPE -> bf16 q/k in [b,h,s,d] layout
//   4. V permute+convert -> bf16 [b,h_kv,s,d]
//   5. WMMA scores (Q.K^T * scale + mask), causal tile skip -> attn_weights
//   6. row softmax in place
//   7. WMMA P.V (causal-clipped K loop) -> bf16 attn heads
//   8. LDS-tiled WMMA GEMM: output projection
// ---------------------------------------------------------------------------

typedef __attribute__((ext_vector_type(16))) __bf16 v16bf;
typedef __attribute__((ext_vector_type(8)))  float  v8f;
typedef unsigned int u32x4 __attribute__((ext_vector_type(4)));

constexpr int B_   = 2;
constexpr int S_   = 2048;
constexpr int HID_ = 3584;
constexpr int NH_  = 16;
constexpr int NKV_ = 8;
constexpr int HD_  = 256;
constexpr int M_    = B_ * S_;      // 4096
constexpr int NQ_   = NH_ * HD_;    // 4096
constexpr int NKVD_ = NKV_ * HD_;   // 2048
constexpr float EPS_     = 1e-6f;
constexpr float SCALING_ = 0.0625f; // 256^-0.5

union alignas(32) U16x16 {
  unsigned short u[16];
  u32x4          q[2];
  v16bf          v;
};

__device__ __forceinline__ unsigned short f2bf(float f) {
  unsigned int u = __float_as_uint(f);
  u += 0x7FFFu + ((u >> 16) & 1u);
  return (unsigned short)(u >> 16);
}

// ---------------------------------------------------------------------------
// Async global -> LDS copy (16 bytes / lane), ASYNCcnt-tracked on gfx1250.
// ROCm 7.2 signature: (V4i AS1* src, V4i AS3* dst, imm offset, imm cpol).
// Guarded: falls back to a synchronous register copy if the builtin is absent.
// ---------------------------------------------------------------------------
#if defined(__gfx1250__) && __has_builtin(__builtin_amdgcn_global_load_async_to_lds_b128)
#define HAVE_ASYNC_LDS 1
#else
#define HAVE_ASYNC_LDS 0
#endif

typedef int i32x4v __attribute__((vector_size(16)));
typedef __attribute__((address_space(1))) i32x4v as1_i32x4;
typedef __attribute__((address_space(3))) i32x4v as3_i32x4;

__device__ __forceinline__ void cp16_to_lds(const unsigned short* g, unsigned short* l) {
#if HAVE_ASYNC_LDS
  // Flat LDS addresses carry the LDS byte offset in addr[31:0] (aperture is
  // addr[63:32]), so truncation yields a valid AS(3) pointer.
  __builtin_amdgcn_global_load_async_to_lds_b128(
      (as1_i32x4*)(unsigned long long)g,
      (as3_i32x4*)(unsigned int)(unsigned long long)l,
      0, 0);
#else
  *(u32x4*)l = *(const u32x4*)g;
#endif
}

__device__ __forceinline__ void wait_async_lds() {
#if HAVE_ASYNC_LDS
#if __has_builtin(__builtin_amdgcn_s_wait_asynccnt)
  __builtin_amdgcn_s_wait_asynccnt(0);
#else
  asm volatile("s_wait_asynccnt 0" ::: "memory");
#endif
#endif
}

// ---------------------------------------------------------------------------
// LDS-tiled WMMA GEMM: C[M,N] (f32) = A[M,K] (bf16) * W[N,K]^T (bf16)
// Block = 256 threads (8 waves) computes a 128x64 macro-tile.
// Waves arranged 4(M) x 2(N); each wave owns 32x32 = 2x2 WMMA tiles.
// K is consumed in 32-wide slabs, double-buffered in LDS with async fills.
// LDS row stride = 40 bf16 (80 B): 16 fragment rows hit disjoint bank groups.
// ---------------------------------------------------------------------------
constexpr int BM = 128, BN = 64, BK = 32;
constexpr int LDA = BK + 8; // padded row stride (elements); 80 B, 16B-aligned

__global__ void gemm_bf16_wmma_lds(const unsigned short* __restrict__ A,
                                   const unsigned short* __restrict__ W,
                                   float* __restrict__ C, int M, int N, int K) {
  __shared__ unsigned short As[2][BM * LDA]; // 2 * 128*40*2B = 20.0 KB
  __shared__ unsigned short Bs[2][BN * LDA]; // 2 *  64*40*2B = 10.0 KB

  const int tid  = threadIdx.x;
  const int lane = tid & 31;
  const int wave = tid >> 5;
  const int nbn  = N / BN;
  const int bm0  = (blockIdx.x / nbn) * BM;
  const int bn0  = (blockIdx.x % nbn) * BN;
  const int wm0  = (wave & 3) * 32;
  const int wn0  = (wave >> 2) * 32;
  const int l16  = lane & 15, hi = lane >> 4;
  const int kgA  = hi * 8, kgB = hi * 8;

  // Fill mapping: 16B chunks of a [rows][BK] slab; 4 chunks per row.
  const int arow = tid >> 2;            // 0..63 (second A chunk: +64 rows)
  const int akc  = (tid & 3) * 8;       // k offset within slab
  const unsigned short* Ag0 = A + (size_t)(bm0 + arow) * K + akc;
  const unsigned short* Ag1 = A + (size_t)(bm0 + arow + 64) * K + akc;
  const unsigned short* Bg  = W + (size_t)(bn0 + arow) * K + akc;

  auto fill = [&](int buf, int k0) {
    cp16_to_lds(Ag0 + k0, &As[buf][arow * LDA + akc]);
    cp16_to_lds(Ag1 + k0, &As[buf][(arow + 64) * LDA + akc]);
    cp16_to_lds(Bg + k0, &Bs[buf][arow * LDA + akc]);
  };

  v8f acc[2][2];
#pragma unroll
  for (int i = 0; i < 2; ++i)
#pragma unroll
    for (int j = 0; j < 2; ++j)
#pragma unroll
      for (int r = 0; r < 8; ++r) acc[i][j][r] = 0.0f;

  fill(0, 0);
  wait_async_lds();
  __syncthreads();

  const int nk = K / BK;
  for (int ks = 0; ks < nk; ++ks) {
    const int cur = ks & 1;
    if (ks + 1 < nk) fill(cur ^ 1, (ks + 1) * BK);

    U16x16 af[2], bf2[2];
#pragma unroll
    for (int i = 0; i < 2; ++i) {
      const int row = (wm0 + i * 16 + l16) * LDA;
      af[i].q[0] = *(const u32x4*)&As[cur][row + kgA];
      af[i].q[1] = *(const u32x4*)&As[cur][row + kgA + 16];
      const int col = (wn0 + i * 16 + l16) * LDA;
      bf2[i].q[0] = *(const u32x4*)&Bs[cur][col + kgB];
      bf2[i].q[1] = *(const u32x4*)&Bs[cur][col + kgB + 16];
    }
#pragma unroll
    for (int i = 0; i < 2; ++i)
#pragma unroll
      for (int j = 0; j < 2; ++j)
        acc[i][j] = __builtin_amdgcn_wmma_f32_16x16x32_bf16(
            false, af[i].v, false, bf2[j].v, (short)0, acc[i][j], false, false);

    wait_async_lds();
    __syncthreads();
  }

#pragma unroll
  for (int i = 0; i < 2; ++i) {
    const int m = bm0 + wm0 + i * 16 + hi * 8;
#pragma unroll
    for (int j = 0; j < 2; ++j) {
      const int n = bn0 + wn0 + j * 16 + l16;
#pragma unroll
      for (int r = 0; r < 8; ++r)
        C[(size_t)(m + r) * N + n] = acc[i][j][r];
    }
  }
}

// ---------------------------------------------------------------------------
__global__ void cvt_f32_bf16(const float* __restrict__ in,
                             unsigned short* __restrict__ out, size_t n) {
  size_t i = (size_t)blockIdx.x * blockDim.x + threadIdx.x;
  const size_t stride = (size_t)gridDim.x * blockDim.x;
  for (; i < n; i += stride) out[i] = f2bf(in[i]);
}

// ---------------------------------------------------------------------------
// Fused RMSNorm + RoPE. One 256-thread block per (b,s,head); thread per d.
// In: X [b,s,heads,HD] fp32.  Out: [b,h,s,HD] bf16.
// ---------------------------------------------------------------------------
__global__ void rmsnorm_rope_kernel(const float* __restrict__ X,
                                    const float* __restrict__ w,
                                    const float* __restrict__ cosb,
                                    const float* __restrict__ sinb,
                                    unsigned short* __restrict__ out,
                                    int nheads) {
  const int d = threadIdx.x;
  int blk = blockIdx.x;
  const int h = blk % nheads; blk /= nheads;
  const int s = blk % S_;
  const int b = blk / S_;

  __shared__ float red[256];
  __shared__ float xn[256];

  const float x = X[(((size_t)(b * S_ + s)) * nheads + h) * HD_ + d];
  red[d] = x * x;
  __syncthreads();
#pragma unroll
  for (int t = 128; t > 0; t >>= 1) {
    if (d < t) red[d] += red[d + t];
    __syncthreads();
  }
  const float rs = rsqrtf(red[0] * (1.0f / HD_) + EPS_);
  const float v  = x * rs * (1.0f + w[d]);
  xn[d] = v;
  __syncthreads();
  const float rot = (d < 128) ? -xn[d + 128] : xn[d - 128];
  const size_t ci = ((size_t)(b * S_ + s)) * HD_ + d;
  const float val = v * cosb[ci] + rot * sinb[ci];
  out[(((size_t)(b * nheads + h)) * S_ + s) * HD_ + d] = f2bf(val);
}

// ---------------------------------------------------------------------------
// V: [b,s,h_kv,d] fp32 -> [b,h_kv,s,d] bf16
// ---------------------------------------------------------------------------
__global__ void permute_v_bf16(const float* __restrict__ Vf,
                               unsigned short* __restrict__ vr) {
  size_t i = (size_t)blockIdx.x * blockDim.x + threadIdx.x;
  const size_t n = (size_t)M_ * NKVD_;
  const size_t stride = (size_t)gridDim.x * blockDim.x;
  for (; i < n; i += stride) {
    const int d = (int)(i % HD_);
    size_t r = i / HD_;
    const int h = (int)(r % NKV_); r /= NKV_;
    const int s = (int)(r % S_);
    const int b = (int)(r / S_);
    vr[(((size_t)(b * NKV_ + h)) * S_ + s) * HD_ + d] = f2bf(Vf[i]);
  }
}

// ---------------------------------------------------------------------------
// Scores: attw[b,h,q,k] = (Q . K) * SCALING + mask.  One 16x16 tile / wave.
// Fully-masked causal tiles (k0 > q0+15) skip the WMMA loop entirely and
// just forward the mask (softmax needs only the NEG values there).
// ---------------------------------------------------------------------------
__global__ void scores_wmma(const unsigned short* __restrict__ qr,
                            const unsigned short* __restrict__ kr,
                            const float* __restrict__ mask,
                            float* __restrict__ attw) {
  const int lane = threadIdx.x & 31;
  const int wave = threadIdx.x >> 5;
  int tile = blockIdx.x * 8 + wave;
  const int nkt = S_ >> 4;
  const int kt = tile % nkt; tile /= nkt;
  const int qt = tile % nkt; tile /= nkt;
  const int h  = tile % NH_;
  const int b  = tile / NH_;
  const int hk = h >> 1;                     // GQA: NH/NKV == 2
  const int q0 = qt << 4, k0 = kt << 4;
  const int l16 = lane & 15, hi = lane >> 4;

  float* orow = attw + (((size_t)(b * NH_ + h)) * S_);

  if (k0 > q0 + 15) {                        // fully masked tile
#pragma unroll
    for (int r = 0; r < 8; ++r) {
      const int q = q0 + r + hi * 8;
      const int k = k0 + l16;
      orow[(size_t)q * S_ + k] = mask[((size_t)b * S_ + q) * S_ + k];
    }
    return;
  }

  const unsigned short* Aq = qr + (((size_t)(b * NH_  + h )) * S_ + q0 + l16) * HD_;
  const unsigned short* Bk = kr + (((size_t)(b * NKV_ + hk)) * S_ + k0 + l16) * HD_;
  const int kgA = hi * 8, kgB = hi * 8;

  v8f acc;
#pragma unroll
  for (int i = 0; i < 8; ++i) acc[i] = 0.0f;

#pragma unroll
  for (int d0 = 0; d0 < HD_; d0 += 32) {
    U16x16 a, bm;
    a.q[0]  = *(const u32x4*)(Aq + d0 + kgA);
    a.q[1]  = *(const u32x4*)(Aq + d0 + kgA + 16);
    bm.q[0] = *(const u32x4*)(Bk + d0 + kgB);
    bm.q[1] = *(const u32x4*)(Bk + d0 + kgB + 16);
    acc = __builtin_amdgcn_wmma_f32_16x16x32_bf16(false, a.v, false, bm.v,
                                                  (short)0, acc, false, false);
  }
#pragma unroll
  for (int r = 0; r < 8; ++r) {
    const int q = q0 + r + hi * 8;
    const int k = k0 + l16;
    orow[(size_t)q * S_ + k] =
        acc[r] * SCALING_ + mask[((size_t)b * S_ + q) * S_ + k];
  }
}

// ---------------------------------------------------------------------------
// Row softmax in place (one block per (b,h,q) row of S_=2048, 8 elems/thread)
// ---------------------------------------------------------------------------
__global__ void softmax_rows(float* __restrict__ attw) {
  const int tid = threadIdx.x;
  float* p = attw + (size_t)blockIdx.x * S_;
  __shared__ float red[256];
  float vals[8];
  float mx = -3.0e38f;
#pragma unroll
  for (int i = 0; i < 8; ++i) { vals[i] = p[tid + (i << 8)]; mx = fmaxf(mx, vals[i]); }
  red[tid] = mx; __syncthreads();
#pragma unroll
  for (int t = 128; t > 0; t >>= 1) {
    if (tid < t) red[tid] = fmaxf(red[tid], red[tid + t]);
    __syncthreads();
  }
  mx = red[0]; __syncthreads();
  float sum = 0.0f;
#pragma unroll
  for (int i = 0; i < 8; ++i) { vals[i] = __expf(vals[i] - mx); sum += vals[i]; }
  red[tid] = sum; __syncthreads();
#pragma unroll
  for (int t = 128; t > 0; t >>= 1) {
    if (tid < t) red[tid] += red[tid + t];
    __syncthreads();
  }
  const float inv = 1.0f / red[0];
#pragma unroll
  for (int i = 0; i < 8; ++i) p[tid + (i << 8)] = vals[i] * inv;
}

// ---------------------------------------------------------------------------
// P.V: AO[b,q,h*HD+d] = sum_k P[b,h,q,k] * V[b,h/2,k,d]   (bf16 out)
// ---------------------------------------------------------------------------
__global__ void pv_wmma(const float* __restrict__ attw,
                        const unsigned short* __restrict__ vr,
                        unsigned short* __restrict__ aob) {
  const int lane = threadIdx.x & 31;
  const int wave = threadIdx.x >> 5;
  int tile = blockIdx.x * 8 + wave;
  const int ndt = HD_ >> 4;                  // 16
  const int dt = tile % ndt; tile /= ndt;
  const int qt = tile % (S_ >> 4); tile /= (S_ >> 4);
  const int h  = tile % NH_;
  const int b  = tile / NH_;
  const int hk = h >> 1;
  const int q0 = qt << 4, d0 = dt << 4;
  const int l16 = lane & 15, hi = lane >> 4;

  const float* Pq = attw + (((size_t)(b * NH_ + h)) * S_ + q0 + l16) * S_;
  const unsigned short* Vb =
      vr + ((size_t)(b * NKV_ + hk)) * S_ * HD_ + (d0 + l16);
  const int kgA = hi * 16, kgB = hi * 16;

  int kmax = (q0 + 16 + 31) & ~31;           // causal clip
  if (kmax > S_) kmax = S_;

  v8f acc;
#pragma unroll
  for (int i = 0; i < 8; ++i) acc[i] = 0.0f;

  for (int k0 = 0; k0 < kmax; k0 += 32) {
    U16x16 a, bm;
#pragma unroll
    for (int e = 0; e < 8; ++e) {
      a.u[e]     = f2bf(Pq[k0 + kgA + e]);
      a.u[8 + e] = f2bf(Pq[k0 + kgA + 8 + e]);
    }
#pragma unroll
    for (int e = 0; e < 16; ++e)
      bm.u[e] = Vb[(size_t)(k0 + kgB + e) * HD_];
    acc = __builtin_amdgcn_wmma_f32_16x16x32_bf16(false, a.v, false, bm.v,
                                                  (short)0, acc, false, false);
  }
#pragma unroll
  for (int r = 0; r < 8; ++r) {
    const int q = q0 + r + hi * 8;
    aob[((size_t)(b * S_ + q)) * NQ_ + h * HD_ + (d0 + l16)] = f2bf(acc[r]);
  }
}

// ---------------------------------------------------------------------------
extern "C" void kernel_launch(void* const* d_in, const int* in_sizes, int n_in,
                              void* d_out, int out_size, void* d_ws, size_t ws_size,
                              hipStream_t stream) {
  const float* hs   = (const float*)d_in[0];
  const float* cosb = (const float*)d_in[1];
  const float* sinb = (const float*)d_in[2];
  const float* mask = (const float*)d_in[3];
  const float* Wq   = (const float*)d_in[4];
  const float* Wk   = (const float*)d_in[5];
  const float* Wv   = (const float*)d_in[6];
  const float* Wo   = (const float*)d_in[7];
  const float* qnw  = (const float*)d_in[8];
  const float* knw  = (const float*)d_in[9];

  float* out_attn = (float*)d_out;                        // [B,S,HID]
  float* attw     = out_attn + (size_t)M_ * HID_;         // [B,NH,S,S]

  char* w = (char*)d_ws;
  auto take = [&](size_t bytes) { char* p = w; w += (bytes + 255) & ~(size_t)255; return p; };
  unsigned short* hsb = (unsigned short*)take((size_t)M_ * HID_ * 2);
  unsigned short* wqb = (unsigned short*)take((size_t)NQ_ * HID_ * 2);
  unsigned short* wkb = (unsigned short*)take((size_t)NKVD_ * HID_ * 2);
  unsigned short* wvb = (unsigned short*)take((size_t)NKVD_ * HID_ * 2);
  unsigned short* wob = (unsigned short*)take((size_t)HID_ * NQ_ * 2);
  float* Qf           = (float*)take((size_t)M_ * NQ_ * 4);
  float* Kf           = (float*)take((size_t)M_ * NKVD_ * 4);
  float* Vf           = (float*)take((size_t)M_ * NKVD_ * 4);
  unsigned short* qr  = (unsigned short*)take((size_t)B_ * NH_ * S_ * HD_ * 2);
  unsigned short* kr  = (unsigned short*)take((size_t)B_ * NKV_ * S_ * HD_ * 2);
  unsigned short* vr  = (unsigned short*)take((size_t)B_ * NKV_ * S_ * HD_ * 2);
  unsigned short* aob = (unsigned short*)take((size_t)M_ * NQ_ * 2);

  auto cvt = [&](const float* src, unsigned short* dst, size_t n) {
    int blocks = (int)((n + 2047) / 2048);
    cvt_f32_bf16<<<blocks, 256, 0, stream>>>(src, dst, n);
  };

  cvt(hs, hsb, (size_t)M_ * HID_);
  cvt(Wq, wqb, (size_t)NQ_ * HID_);
  cvt(Wk, wkb, (size_t)NKVD_ * HID_);
  cvt(Wv, wvb, (size_t)NKVD_ * HID_);
  cvt(Wo, wob, (size_t)HID_ * NQ_);

  auto gemm = [&](const unsigned short* A, const unsigned short* Wm, float* C,
                  int M, int N, int K) {
    int blocks = (M / BM) * (N / BN);
    gemm_bf16_wmma_lds<<<blocks, 256, 0, stream>>>(A, Wm, C, M, N, K);
  };
  gemm(hsb, wqb, Qf, M_, NQ_,   HID_);
  gemm(hsb, wkb, Kf, M_, NKVD_, HID_);
  gemm(hsb, wvb, Vf, M_, NKVD_, HID_);

  rmsnorm_rope_kernel<<<B_ * S_ * NH_,  256, 0, stream>>>(Qf, qnw, cosb, sinb, qr, NH_);
  rmsnorm_rope_kernel<<<B_ * S_ * NKV_, 256, 0, stream>>>(Kf, knw, cosb, sinb, kr, NKV_);

  permute_v_bf16<<<(int)(((size_t)M_ * NKVD_ + 2047) / 2048), 256, 0, stream>>>(Vf, vr);

  {
    int tiles = B_ * NH_ * (S_ >> 4) * (S_ >> 4);
    scores_wmma<<<tiles / 8, 256, 0, stream>>>(qr, kr, mask, attw);
  }

  softmax_rows<<<B_ * NH_ * S_, 256, 0, stream>>>(attw);

  {
    int tiles = B_ * NH_ * (S_ >> 4) * (HD_ >> 4);
    pv_wmma<<<tiles / 8, 256, 0, stream>>>(attw, vr, aob);
  }

  gemm(aob, wob, out_attn, M_, HID_, NQ_);
}
